// GraphMultisetTransformer_63136019251612
// MI455X (gfx1250) — matrix operand order repaired and checked by reading, not compiled, not run on hardware
//
#include <hip/hip_runtime.h>
#include <math.h>

typedef __attribute__((ext_vector_type(16))) _Float16 v16h;
typedef __attribute__((ext_vector_type(8)))  float    v8f;

static constexpr int NN   = 20000;   // nodes
static constexpr int GG   = 200;     // graphs
static constexpr int MAXN = 100;     // nodes per graph (batch = arange//100 -> exact)
static constexpr int EE   = 320000;  // edges
static constexpr int HID  = 128;
static constexpr int NSEED = 25;

// ---------------------------------------------------------------- elementwise
__global__ void k_fill(float* __restrict__ p, float v, int n) {
  int i = blockIdx.x * blockDim.x + threadIdx.x;
  if (i < n) p[i] = v;
}

__global__ void k_deg_edges(const int* __restrict__ dst, float* __restrict__ deg, int e) {
  int i = blockIdx.x * blockDim.x + threadIdx.x;
  if (i < e) atomicAdd(&deg[dst[i]], 1.0f);
}

__global__ void k_rsqrt_deg(float* __restrict__ d, int n) {
  int i = blockIdx.x * blockDim.x + threadIdx.x;
  if (i < n) d[i] = rsqrtf(fmaxf(d[i], 1.0f));
}

__global__ void k_enorm(const int* __restrict__ src, const int* __restrict__ dst,
                        const float* __restrict__ dinv, float* __restrict__ en, int e) {
  int i = blockIdx.x * blockDim.x + threadIdx.x;
  if (i < e) en[i] = dinv[src[i]] * dinv[dst[i]];
}

// transpose + f32->f16 weight prep: W[kin][kout] -> Wt[kout][kin]
__global__ void k_w2h(const float* __restrict__ W, _Float16* __restrict__ Wt,
                      int kin, int kout) {
  int i = blockIdx.x * blockDim.x + threadIdx.x;
  if (i < kin * kout) {
    int ki = i / kout, ko = i % kout;
    Wt[(size_t)ko * kin + ki] = (_Float16)W[i];
  }
}

// --------------------------------------------------------------- WMMA GEMM
// C[M,Kout] = A[M,Kin] @ W[Kin,Kout] (+bias) (+relu), Wt is W transposed in f16.
// One wave per 16x16 output tile, K-loop in steps of 32 via v_wmma_f32_16x16x32_f16.
// OOB lanes are CLAMPED to valid addresses (never masked): garbage only lands in
// D rows/cols that the store guards skip, so all loads are unconditional+vector.
__global__ void __launch_bounds__(256) k_gemm(
    const float* __restrict__ A, const _Float16* __restrict__ Wt,
    const float* __restrict__ bias, float* __restrict__ C,
    int M, int Kin, int Kout, int lda, int ldc, int relu)
{
  int lane = threadIdx.x & 31;
  int wave = blockIdx.x * (blockDim.x >> 5) + (threadIdx.x >> 5);
  int mtiles = (M + 15) >> 4;
  int ntiles = (Kout + 15) >> 4;
  if (wave >= mtiles * ntiles) return;          // wave-uniform exit, EXEC stays all-1s
  int mt = wave / ntiles, nt = wave % ntiles;
  int hs = lane >> 4;                           // half-wave select
  int l  = lane & 15;
  int rowA = mt * 16 + l;
  int colB = nt * 16 + l;
  const float*    Arow = A  + (size_t)(rowA < M ? rowA : M - 1) * lda;
  const _Float16* Brow = Wt + (size_t)(colB < Kout ? colB : Kout - 1) * Kin;

  v8f acc = {};
  for (int k = 0; k < Kin; k += 32) {
    // A 16x32 f16 layout: lanes<16 -> K k..k+15 ; lanes>=16 -> +16
    const float4* pa0 = (const float4*)(Arow + k + hs * 16);
    const float4* pa1 = (const float4*)(Arow + k + 8 + hs * 16);
    float4 f0 = pa0[0], f1 = pa0[1];
    float4 f2 = pa1[0], f3 = pa1[1];
    v16h a;
    a[0]  = (_Float16)f0.x; a[1]  = (_Float16)f0.y;
    a[2]  = (_Float16)f0.z; a[3]  = (_Float16)f0.w;
    a[4]  = (_Float16)f1.x; a[5]  = (_Float16)f1.y;
    a[6]  = (_Float16)f1.z; a[7]  = (_Float16)f1.w;
    a[8]  = (_Float16)f2.x; a[9]  = (_Float16)f2.y;
    a[10] = (_Float16)f2.z; a[11] = (_Float16)f2.w;
    a[12] = (_Float16)f3.x; a[13] = (_Float16)f3.y;
    a[14] = (_Float16)f3.z; a[15] = (_Float16)f3.w;
    // B 32x16 f16 layout: lanes<16 -> K k..k+15 of column lane ; lanes>=16 -> +16
    v16h b = *(const v16h*)(Brow + k + hs * 16);   // 32B-aligned (Kin mult of 32)
    acc = __builtin_amdgcn_wmma_f32_16x16x32_f16(false, a, false, b, (short)0, acc,
                                                 false, false);
  }
  if (colB < Kout) {
    float bv = bias ? bias[colB] : 0.0f;
    int mbase = mt * 16 + hs * 8;                 // C layout: lanes>=16 hold M=r+8
    if (mbase + 8 <= M) {                         // full tile fast path
#pragma unroll
      for (int r = 0; r < 8; ++r) {
        float v = acc[r] + bv;
        if (relu) v = fmaxf(v, 0.0f);
        C[(size_t)(mbase + r) * ldc + colB] = v;
      }
    } else {
#pragma unroll
      for (int r = 0; r < 8; ++r) {
        if (mbase + r < M) {
          float v = acc[r] + bv;
          if (relu) v = fmaxf(v, 0.0f);
          C[(size_t)(mbase + r) * ldc + colB] = v;
        }
      }
    }
  }
}

// --------------------------------------------------------------- GCN pieces
// self-loop term + bias (plain assignment, defines the output buffer)
__global__ void k_gcn_self(const float* __restrict__ h, const float* __restrict__ dinv,
                           const float* __restrict__ bias, float* __restrict__ out,
                           int n, int dim, int ldo, int coff) {
  int i = blockIdx.x * blockDim.x + threadIdx.x;
  if (i < n * dim) {
    int r = i / dim, f = i % dim;
    float di = dinv[r];
    out[(size_t)r * ldo + coff + f] = h[(size_t)r * dim + f] * di * di + bias[f];
  }
}

// edge scatter-add: out[dst] += h[src] * norm  (dominant HBM traffic of the model)
__global__ void k_gcn_edges(const int* __restrict__ src, const int* __restrict__ dst,
                            const float* __restrict__ en, const float* __restrict__ h,
                            float* __restrict__ out, int e, int dim, int ldo, int coff) {
  long i = (long)blockIdx.x * blockDim.x + threadIdx.x;
  if (i < (long)e * dim) {
    int ei = (int)(i / dim), f = (int)(i % dim);
    atomicAdd(&out[(size_t)dst[ei] * ldo + coff + f],
              h[(size_t)src[ei] * dim + f] * en[ei]);
  }
}

__global__ void k_relu_strided(float* __restrict__ p, int rows, int cols, int ld) {
  int i = blockIdx.x * blockDim.x + threadIdx.x;
  if (i < rows * cols) {
    int r = i / cols, c = i % cols;
    float* q = &p[(size_t)r * ld + c];
    *q = fmaxf(*q, 0.0f);
  }
}

// global_add_pool into d_out region
__global__ void k_pool(const int* __restrict__ batch, const float* __restrict__ xcat,
                       float* __restrict__ outp, int coff) {
  int i = blockIdx.x * blockDim.x + threadIdx.x;
  if (i < NN * HID) {
    int r = i / HID, f = i % HID;
    atomicAdd(&outp[(size_t)batch[r] * HID + f], xcat[(size_t)r * 256 + coff + f]);
  }
}

// --------------------------------------------------------------- attention
// One block per (graph, head). Softmax over the QUERY (seed) axis, per key.
// Mask is all-true (every graph has exactly MAXN nodes), so no bias term.
__global__ void __launch_bounds__(256) k_attn(
    const float* __restrict__ Q, const float* __restrict__ Kp, const float* __restrict__ Vp,
    float* __restrict__ O, int nq, int nk, int dv, int dh, float scale, long qgs)
{
  __shared__ float sm[10500];           // max: 25*64 + 100*64 + 25*100 floats
  int g = blockIdx.x, h = blockIdx.y;
  float* sQ = sm;
  float* sK = sm + nq * dh;
  float* sS = sK + nk * dh;
  const float* Qb = Q  + (size_t)g * qgs + (size_t)h * dh;
  const float* Kb = Kp + (size_t)g * nk * dv + (size_t)h * dh;
  const float* Vb = Vp + (size_t)g * nk * dv + (size_t)h * dh;
  int t = threadIdx.x, nthr = blockDim.x;

  for (int i = t; i < nq * dh; i += nthr) sQ[i] = Qb[(size_t)(i / dh) * dv + (i % dh)];
  for (int i = t; i < nk * dh; i += nthr) sK[i] = Kb[(size_t)(i / dh) * dv + (i % dh)];
  __syncthreads();
  for (int i = t; i < nq * nk; i += nthr) {
    int q = i / nk, kk = i % nk;
    float acc = 0.0f;
    for (int d = 0; d < dh; ++d) acc += sQ[q * dh + d] * sK[kk * dh + d];
    sS[i] = acc * scale;
  }
  __syncthreads();
  for (int kk = t; kk < nk; kk += nthr) {        // softmax over q, per key column
    float mx = -3.4e38f;
    for (int q = 0; q < nq; ++q) mx = fmaxf(mx, sS[q * nk + kk]);
    float sum = 0.0f;
    for (int q = 0; q < nq; ++q) {
      float e = __expf(sS[q * nk + kk] - mx);
      sS[q * nk + kk] = e; sum += e;
    }
    float inv = 1.0f / sum;
    for (int q = 0; q < nq; ++q) sS[q * nk + kk] *= inv;
  }
  __syncthreads();
  for (int i = t; i < nq * dh; i += nthr) {
    int q = i / dh, d = i % dh;
    float acc = 0.0f;
    for (int kk = 0; kk < nk; ++kk) acc += sS[q * nk + kk] * Vb[(size_t)kk * dv + d];
    O[(size_t)g * nq * dv + (size_t)q * dv + (size_t)h * dh + d] = acc;
  }
}

// T += Q (Q broadcast when qgs==0)
__global__ void k_addq(float* __restrict__ T, const float* __restrict__ Q,
                       int perG, long qgs, int total) {
  int i = blockIdx.x * blockDim.x + threadIdx.x;
  if (i < total) {
    int g = i / perG, r = i % perG;
    T[i] += Q[(size_t)g * qgs + r];
  }
}

// T += relu(U)
__global__ void k_resrelu(float* __restrict__ T, const float* __restrict__ U, int n) {
  int i = blockIdx.x * blockDim.x + threadIdx.x;
  if (i < n) T[i] += fmaxf(U[i], 0.0f);
}

// GMPool_I with 1 seed: softmax over the (single) query axis == 1 everywhere,
// so O2 = Q2 + sum_k V2[k]. (The K projection is mathematically dead.)
__global__ void k_stage2(const float* __restrict__ q2, const float* __restrict__ V2,
                         float* __restrict__ O2) {
  int i = blockIdx.x * blockDim.x + threadIdx.x;
  if (i < GG * HID) {
    int g = i / HID, f = i % HID;
    float acc = q2[f];
    for (int k = 0; k < NSEED; ++k) acc += V2[((size_t)g * NSEED + k) * HID + f];
    O2[i] = acc;
  }
}

// --------------------------------------------------------------- host driver
extern "C" void kernel_launch(void* const* d_in, const int* in_sizes, int n_in,
                              void* d_out, int out_size, void* d_ws, size_t ws_size,
                              hipStream_t stream) {
  (void)in_sizes; (void)n_in; (void)ws_size;
  const float* x     = (const float*)d_in[0];
  const int*   eidx  = (const int*)d_in[1];
  const int*   esrc  = eidx;
  const int*   edst  = eidx + EE;
  const int*   batch = (const int*)d_in[2];
  const float *W1=(const float*)d_in[3], *b1=(const float*)d_in[4];
  const float *W2=(const float*)d_in[5], *b2=(const float*)d_in[6];
  const float *S0=(const float*)d_in[7];
  const float *Wq0=(const float*)d_in[8],  *bq0=(const float*)d_in[9];
  const float *Wk0=(const float*)d_in[10], *bk0=(const float*)d_in[11];
  const float *Wv0=(const float*)d_in[12], *bv0=(const float*)d_in[13];
  const float *Wo0=(const float*)d_in[14], *bo0=(const float*)d_in[15];
  const float *Wq1=(const float*)d_in[16], *bq1=(const float*)d_in[17];
  const float *Wk1=(const float*)d_in[18], *bk1=(const float*)d_in[19];
  const float *Wv1=(const float*)d_in[20], *bv1=(const float*)d_in[21];
  const float *Wo1=(const float*)d_in[22], *bo1=(const float*)d_in[23];
  const float *S2=(const float*)d_in[24];
  const float *Wq2=(const float*)d_in[25], *bq2=(const float*)d_in[26];
  // Wk2/bk2 (d_in[27..28]) are dead: single-seed softmax-over-query is constant 1.
  const float *Wv2=(const float*)d_in[29], *bv2=(const float*)d_in[30];
  const float *Wo2=(const float*)d_in[31], *bo2=(const float*)d_in[32];
  const float *Wl=(const float*)d_in[33],  *bl=(const float*)d_in[34];
  const float *Wc1=(const float*)d_in[35], *bc1=(const float*)d_in[36];
  const float *Wc2=(const float*)d_in[37], *bc2=(const float*)d_in[38];
  const float *Wc3=(const float*)d_in[39], *bc3=(const float*)d_in[40];

  // bump allocator over d_ws (~96 MB total)
  char* base = (char*)d_ws;
  size_t off = 0;
  auto alloc = [&](size_t elems, size_t esz) -> void* {
    void* p = base + off;
    off += (elems * esz + 255) & ~(size_t)255;
    return p;
  };
  float* dinv = (float*)alloc(NN, 4);
  float* en   = (float*)alloc(EE, 4);
  float* hbuf = (float*)alloc((size_t)NN * 256, 4);   // GEMM temp for GCN layers
  float* xcat = (float*)alloc((size_t)NN * 256, 4);   // cols 0..127 = x1, 128..255 = x2
  float* Kn   = (float*)alloc((size_t)NN * 256, 4);   // dense K (reshape of node array)
  float* Vn   = (float*)alloc((size_t)NN * 256, 4);
  float* q0   = (float*)alloc((size_t)NSEED * 256, 4);
  float* a0   = (float*)alloc((size_t)GG * NSEED * 256, 4);
  float* ub   = (float*)alloc((size_t)GG * NSEED * 256, 4);
  float* q1   = (float*)alloc((size_t)GG * NSEED * 128, 4);
  float* k1b  = (float*)alloc((size_t)GG * NSEED * 128, 4);
  float* v1b  = (float*)alloc((size_t)GG * NSEED * 128, 4);
  float* a1   = (float*)alloc((size_t)GG * NSEED * 128, 4);
  float* q2   = (float*)alloc(128, 4);
  float* o2   = (float*)alloc((size_t)GG * 128, 4);
  float* z1   = (float*)alloc((size_t)GG * 128, 4);
  float* z2   = (float*)alloc((size_t)GG * 128, 4);
  float* z3   = (float*)alloc((size_t)GG * 64, 4);
  _Float16* w1t  = (_Float16*)alloc(128 * 128, 2);
  _Float16* w2t  = (_Float16*)alloc(128 * 128, 2);
  _Float16* wq0t = (_Float16*)alloc(256 * 256, 2);
  _Float16* wk0t = (_Float16*)alloc(256 * 256, 2);
  _Float16* wv0t = (_Float16*)alloc(256 * 256, 2);
  _Float16* wo0t = (_Float16*)alloc(256 * 256, 2);
  _Float16* wq1t = (_Float16*)alloc(256 * 128, 2);
  _Float16* wk1t = (_Float16*)alloc(256 * 128, 2);
  _Float16* wv1t = (_Float16*)alloc(256 * 128, 2);
  _Float16* wo1t = (_Float16*)alloc(128 * 128, 2);
  _Float16* wq2t = (_Float16*)alloc(128 * 128, 2);
  _Float16* wv2t = (_Float16*)alloc(128 * 128, 2);
  _Float16* wo2t = (_Float16*)alloc(128 * 128, 2);
  _Float16* wlt  = (_Float16*)alloc(128 * 128, 2);
  _Float16* wc1t = (_Float16*)alloc(128 * 128, 2);
  _Float16* wc2t = (_Float16*)alloc(128 * 64, 2);
  _Float16* wc3t = (_Float16*)alloc(64 * 10, 2);

  auto fill = [&](float* p, float v, int n) {
    k_fill<<<dim3((n + 255) / 256), dim3(256), 0, stream>>>(p, v, n);
  };
  auto w2h = [&](const float* W, _Float16* Wt, int kin, int kout) {
    int n = kin * kout;
    k_w2h<<<dim3((n + 255) / 256), dim3(256), 0, stream>>>(W, Wt, kin, kout);
  };
  auto gemm = [&](const float* A, const _Float16* Wt, const float* bias, float* C,
                  int M, int Kin, int Kout, int lda, int ldc, int relu) {
    int mt = (M + 15) / 16, ntl = (Kout + 15) / 16;
    int blocks = (mt * ntl + 7) / 8;           // 8 waves / 256-thread block
    k_gemm<<<dim3(blocks), dim3(256), 0, stream>>>(A, Wt, bias, C, M, Kin, Kout,
                                                   lda, ldc, relu);
  };
  auto gcn = [&](const float* Ain, int lda, int dimIn, const _Float16* Wt,
                 const float* bias, int dimOut, float* outp, int ldo, int coff,
                 bool relu) {
    gemm(Ain, Wt, nullptr, hbuf, NN, dimIn, dimOut, lda, dimOut, 0);
    int n1 = NN * dimOut;
    k_gcn_self<<<dim3((n1 + 255) / 256), 256, 0, stream>>>(hbuf, dinv, bias, outp,
                                                           NN, dimOut, ldo, coff);
    long ne = (long)EE * dimOut;
    k_gcn_edges<<<dim3((int)((ne + 255) / 256)), 256, 0, stream>>>(
        esrc, edst, en, hbuf, outp, EE, dimOut, ldo, coff);
    if (relu)
      k_relu_strided<<<dim3((n1 + 255) / 256), 256, 0, stream>>>(outp + coff, NN,
                                                                 dimOut, ldo);
  };

  // --- degrees / norms (shared by all 4 GCNs) ---
  fill(dinv, 1.0f, NN);                                  // self-loop
  k_deg_edges<<<dim3((EE + 255) / 256), 256, 0, stream>>>(edst, dinv, EE);
  k_rsqrt_deg<<<dim3((NN + 255) / 256), 256, 0, stream>>>(dinv, NN);
  k_enorm<<<dim3((EE + 255) / 256), 256, 0, stream>>>(esrc, edst, dinv, en, EE);

  // --- weight prep (transpose + f16) ---
  w2h(W1, w1t, 128, 128);   w2h(W2, w2t, 128, 128);
  w2h(Wq0, wq0t, 256, 256); w2h(Wk0, wk0t, 256, 256);
  w2h(Wv0, wv0t, 256, 256); w2h(Wo0, wo0t, 256, 256);
  w2h(Wq1, wq1t, 256, 128); w2h(Wk1, wk1t, 256, 128); w2h(Wv1, wv1t, 256, 128);
  w2h(Wo1, wo1t, 128, 128);
  w2h(Wq2, wq2t, 128, 128); w2h(Wv2, wv2t, 128, 128); w2h(Wo2, wo2t, 128, 128);
  w2h(Wl, wlt, 128, 128);   w2h(Wc1, wc1t, 128, 128);
  w2h(Wc2, wc2t, 128, 64);  w2h(Wc3, wc3t, 64, 10);

  // --- GCN1 -> x1 (xcat cols 0..127), GCN2 -> x2 (cols 128..255) ---
  gcn(x, 128, 128, w1t, b1, 128, xcat, 256, 0, true);
  gcn(xcat, 256, 128, w2t, b2, 128, xcat, 256, 128, true);

  // --- pools p1/p2 into d_out ---
  fill((float*)d_out, 0.0f, out_size);
  float* p1 = (float*)d_out + GG * 10;
  float* p2 = p1 + GG * HID;
  k_pool<<<dim3((NN * HID + 255) / 256), 256, 0, stream>>>(batch, xcat, p1, 0);
  k_pool<<<dim3((NN * HID + 255) / 256), 256, 0, stream>>>(batch, xcat, p2, 128);

  // --- K/V GCNs on xcat (dense form is a pure reshape: 100 nodes per graph) ---
  gcn(xcat, 256, 256, wk0t, bk0, 256, Kn, 256, 0, false);
  gcn(xcat, 256, 256, wv0t, bv0, 256, Vn, 256, 0, false);

  // --- GMPool_G (25 seeds, broadcast Q) ---
  gemm(S0, wq0t, bq0, q0, NSEED, 256, 256, 256, 256, 0);
  k_attn<<<dim3(GG, 4), 256, 0, stream>>>(q0, Kn, Vn, a0, NSEED, MAXN, 256, 64,
                                          1.0f / 16.0f, 0);
  k_addq<<<dim3((GG * NSEED * 256 + 255) / 256), 256, 0, stream>>>(
      a0, q0, NSEED * 256, 0, GG * NSEED * 256);
  gemm(a0, wo0t, bo0, ub, GG * NSEED, 256, 256, 256, 256, 0);
  k_resrelu<<<dim3((GG * NSEED * 256 + 255) / 256), 256, 0, stream>>>(
      a0, ub, GG * NSEED * 256);                          // a0 = O0

  // --- SAB ---
  gemm(a0, wq1t, bq1, q1, GG * NSEED, 256, 128, 256, 128, 0);
  gemm(a0, wk1t, bk1, k1b, GG * NSEED, 256, 128, 256, 128, 0);
  gemm(a0, wv1t, bv1, v1b, GG * NSEED, 256, 128, 256, 128, 0);
  k_attn<<<dim3(GG, 4), 256, 0, stream>>>(q1, k1b, v1b, a1, NSEED, NSEED, 128, 32,
                                          1.0f / sqrtf(128.0f), (long)NSEED * 128);
  k_addq<<<dim3((GG * NSEED * 128 + 255) / 256), 256, 0, stream>>>(
      a1, q1, NSEED * 128, (long)NSEED * 128, GG * NSEED * 128);
  gemm(a1, wo1t, bo1, ub, GG * NSEED, 128, 128, 128, 128, 0);
  k_resrelu<<<dim3((GG * NSEED * 128 + 255) / 256), 256, 0, stream>>>(
      a1, ub, GG * NSEED * 128);                          // a1 = O1

  // --- GMPool_I (1 seed: attention == uniform 1 over the query axis) ---
  gemm(S2, wq2t, bq2, q2, 1, 128, 128, 128, 128, 0);
  gemm(a1, wv2t, bv2, k1b, GG * NSEED, 128, 128, 128, 128, 0);   // reuse k1b as V2
  k_stage2<<<dim3((GG * HID + 255) / 256), 256, 0, stream>>>(q2, k1b, o2);
  gemm(o2, wo2t, bo2, ub, GG, 128, 128, 128, 128, 0);
  k_resrelu<<<dim3((GG * HID + 255) / 256), 256, 0, stream>>>(o2, ub, GG * HID);

  // --- head MLP -> logits at d_out[0..1999] ---
  gemm(o2, wlt, bl, z1, GG, 128, 128, 128, 128, 0);
  gemm(z1, wc1t, bc1, z2, GG, 128, 128, 128, 128, 1);
  gemm(z2, wc2t, bc2, z3, GG, 128, 64, 128, 64, 1);
  gemm(z3, wc3t, bc3, (float*)d_out, GG, 64, 10, 64, 10, 0);
}
